// SparseMoELayer_79293686218887
// MI455X (gfx1250) — matrix operand order repaired and checked by reading, not compiled, not run on hardware
//
#include <hip/hip_runtime.h>
#include <hip/hip_bf16.h>

#define B_TOK   8192
#define D_INX   1024
#define D_OUTX  1024
#define NEXP    16
#define KSEL    2
#define HDIM    64
#define CAP     (B_TOK * KSEL)   // worst-case tokens per expert bucket
#define TM      128              // tokens per expert tile
#define THREADS 256              // 8 waves (wave32)
#define TILES_MAX (CAP / TM)     // 128

typedef __attribute__((ext_vector_type(16))) __bf16 v16bf;
typedef __attribute__((ext_vector_type(8)))  __bf16 bf16x8;
typedef __attribute__((ext_vector_type(4)))  __bf16 bf16x4;
typedef __attribute__((ext_vector_type(8)))  float  v8f;

union ABFrag { v16bf v; bf16x8 h[2]; };

// CDNA5 async global->LDS copy (ASYNCcnt-tracked, no VGPR data path).
// LDS destination address comes from the lds_off VGPR; generic shared pointers carry the
// wave-relative LDS byte offset in their low 32 bits.
__device__ __forceinline__ void async_copy_b128(uint32_t lds_off, const void* gaddr) {
    asm volatile("global_load_async_to_lds_b128 %0, %1, off"
                 :: "v"(lds_off), "v"(gaddr) : "memory");
}
__device__ __forceinline__ void wait_async0() {
    asm volatile("s_wait_asynccnt 0x0" ::: "memory");
}
__device__ __forceinline__ uint32_t lds_off32(const void* p) {
    return (uint32_t)(uintptr_t)p;
}

// ---------------------------------------------------------------------------
// x (f32) -> bf16, coalesced. 4 elements / thread.
// ---------------------------------------------------------------------------
__global__ void xcvt_kernel(const float* __restrict__ x, __bf16* __restrict__ xb)
{
    size_t i = (size_t)blockIdx.x * blockDim.x + threadIdx.x;
    float4 v = ((const float4*)x)[i];
    bf16x4 o = { (__bf16)v.x, (__bf16)v.y, (__bf16)v.z, (__bf16)v.w };
    ((bf16x4*)xb)[i] = o;
}

// ---------------------------------------------------------------------------
// Weight prep: W1[e][d][h] -> W1t[e][h][d] (bf16), W2[e][h][o] -> W2t[e][o][h]
// (bf16) via 64x64 LDS tile transpose. Rows of W1t/W2t are contiguous in k,
// so the expert kernel can async-DMA them straight into its [n][k] LDS layout.
// grid = (16 tiles, NEXP, 2)
// ---------------------------------------------------------------------------
__global__ void prep_weights_kernel(const float* __restrict__ W1,
                                    const float* __restrict__ W2,
                                    __bf16* __restrict__ W1t,
                                    __bf16* __restrict__ W2t)
{
    __shared__ float t[64][65];
    const int e  = blockIdx.y;
    const int tl = blockIdx.x;          // k-tile (W1) / o-tile (W2)
    const int which = blockIdx.z;
    const int rr = threadIdx.x >> 2;          // 0..63
    const int c0 = (threadIdx.x & 3) * 16;    // 0..48

    const float* src = (which == 0)
        ? W1 + ((size_t)e * D_INX + (size_t)tl * 64 + rr) * HDIM + 0
        : W2 + ((size_t)e * HDIM + rr) * D_OUTX + (size_t)tl * 64;
#pragma unroll
    for (int j = 0; j < 4; ++j) {
        float4 v = ((const float4*)src)[c0 / 4 + j];
        t[rr][c0 + 4*j + 0] = v.x; t[rr][c0 + 4*j + 1] = v.y;
        t[rr][c0 + 4*j + 2] = v.z; t[rr][c0 + 4*j + 3] = v.w;
    }
    __syncthreads();

    const int nn = rr;                  // output row within tile
    __bf16* dst = (which == 0)
        ? W1t + ((size_t)e * HDIM + nn) * D_INX + (size_t)tl * 64 + c0
        : W2t + ((size_t)e * D_OUTX + (size_t)tl * 64 + nn) * HDIM + c0;
#pragma unroll
    for (int j = 0; j < 16; ++j)
        dst[j] = (__bf16)t[c0 + j][nn];
}

// ---------------------------------------------------------------------------
// Gating: logits = x@Wg + bg, top-2 (first-index tie-break), renormalized
// weights (softmax Z cancels). Appends (token, weight) into expert buckets.
// ---------------------------------------------------------------------------
__global__ void moe_gate_kernel(const float* __restrict__ x,
                                const float* __restrict__ Wg,
                                const float* __restrict__ bg,
                                int*   __restrict__ cnt,
                                int*   __restrict__ btok,
                                float* __restrict__ bw)
{
    int b = blockIdx.x * blockDim.x + threadIdx.x;
    if (b >= B_TOK) return;

    float acc[NEXP];
#pragma unroll
    for (int e = 0; e < NEXP; ++e) acc[e] = bg[e];

    const float4* xr = (const float4*)(x + (size_t)b * D_INX);
    for (int d4 = 0; d4 < D_INX / 4; ++d4) {
        float4 xv = xr[d4];
        const float* wr = Wg + (size_t)d4 * 4 * NEXP;
#pragma unroll
        for (int e = 0; e < NEXP; ++e)
            acc[e] += xv.x * wr[e] + xv.y * wr[NEXP + e] +
                      xv.z * wr[2 * NEXP + e] + xv.w * wr[3 * NEXP + e];
    }

    int e0 = 0; float v0 = acc[0];
#pragma unroll
    for (int e = 1; e < NEXP; ++e) if (acc[e] > v0) { v0 = acc[e]; e0 = e; }
    int e1 = -1; float v1 = -3.4e38f;
#pragma unroll
    for (int e = 0; e < NEXP; ++e)
        if (e != e0 && acc[e] > v1) { v1 = acc[e]; e1 = e; }

    float r  = __expf(v1 - v0);
    float w0 = 1.0f / (1.0f + r);
    float w1 = r    / (1.0f + r);

    int s0 = atomicAdd(&cnt[e0], 1);
    btok[e0 * CAP + s0] = b;  bw[e0 * CAP + s0] = w0;
    int s1 = atomicAdd(&cnt[e1], 1);
    btok[e1 * CAP + s1] = b;  bw[e1 * CAP + s1] = w1;
}

// ---------------------------------------------------------------------------
// Fused per-expert MLP over a 128-token tile:
//   h = relu(Xg @ W1[e] + b1[e]);  y = h @ W2[e] + b2[e];  out += gw * y
// bf16 WMMA, f32 accumulate. All staging is async global->LDS DMA of
// pre-converted bf16 data in the final LDS layout (no VALU, no transpose).
// ---------------------------------------------------------------------------
__global__ __launch_bounds__(THREADS, 1)
void moe_expert_kernel(const __bf16* __restrict__ xb,
                       const __bf16* __restrict__ W1t,   // [e][h][d]
                       const __bf16* __restrict__ W2t,   // [e][o][h]
                       const float*  __restrict__ b1,
                       const float*  __restrict__ b2,
                       const int*    __restrict__ cnt,
                       const int*    __restrict__ btok,
                       const float*  __restrict__ bw,
                       float* __restrict__ out)
{
    const int e     = blockIdx.y;
    const int tile  = blockIdx.x;
    const int n_e   = cnt[e];
    const int start = tile * TM;
    if (start >= n_e) return;   // uniform exit: EXEC all-1s at every WMMA

    // Padded strides (multiples of 8 bf16): 16B-aligned b128 access and
    // bank-conflict-free fragment reads across 16 consecutive rows.
    __shared__ __align__(16) __bf16 s_x [2][TM   * 40];  // [row][k0..31]
    __shared__ __align__(16) __bf16 s_w1[2][HDIM * 40];  // [n][k0..31]
    __shared__ __align__(16) __bf16 s_h [TM * 72];       // [row][0..63]
    __shared__ __align__(16) __bf16 s_w2[2][128  * 72];  // [n][k0..63]

    const int tid   = threadIdx.x;
    const int lane  = tid & 31;
    const int wid   = tid >> 5;
    const int half  = lane >> 4;
    const int l16   = lane & 15;
    const int mbase = wid * 16;

    // Staging row owned by this thread (x: 2 threads/row, 16 bf16 each)
    const int  xrow   = tid >> 1;
    const bool xvalid = (start + xrow) < n_e;
    const int  xtok   = xvalid ? btok[e * CAP + start + xrow] : 0; // tail rows
    // copy token 0 (valid memory); their outputs are masked by gw==0 below.

    int   tokr[8];
    float gwr[8];
#pragma unroll
    for (int r2 = 0; r2 < 8; ++r2) {
        int gi = start + mbase + 8 * half + r2;
        if (gi < n_e) { tokr[r2] = btok[e * CAP + gi]; gwr[r2] = bw[e * CAP + gi]; }
        else          { tokr[r2] = 0;                  gwr[r2] = 0.0f; }
    }

    // ---- async staging: gathered x rows + W1t chunk -----------------------
    auto stage_xw = [&](int buf, int kc) {
        {   // x: 128 rows x 32 bf16; this thread: 16 bf16 = 2 x b128
            const int c0 = (tid & 1) * 16;
            uint32_t dst = lds_off32(&s_x[buf][xrow * 40 + c0]);
            const __bf16* src = xb + (size_t)xtok * D_INX + kc * 32 + c0;
            async_copy_b128(dst,      src);
            async_copy_b128(dst + 16, src + 8);
        }
        {   // W1t: 64 rows x 32 bf16; this thread: 8 bf16 = 1 x b128
            const int n    = tid >> 2;
            const int koff = (tid & 3) * 8;
            uint32_t dst = lds_off32(&s_w1[buf][n * 40 + koff]);
            const __bf16* src = W1t + ((size_t)e * HDIM + n) * D_INX + kc * 32 + koff;
            async_copy_b128(dst, src);
        }
    };

    // ---- stage 1 compute: 4 N-tiles x K=32 per chunk ----------------------
    v8f acc1[4];
#pragma unroll
    for (int t = 0; t < 4; ++t) acc1[t] = (v8f)(0.0f);

    auto compute1 = [&](int buf) {
        ABFrag a;   // ISA 16-bit A 16x32: lane half selects K {0-7,16-23}/{8-15,24-31}
        const __bf16* xr = &s_x[buf][(mbase + l16) * 40];
        a.h[0] = *(const bf16x8*)(xr + 8 * half);
        a.h[1] = *(const bf16x8*)(xr + 16 + 8 * half);
#pragma unroll
        for (int nt = 0; nt < 4; ++nt) {
            ABFrag b;   // B 32x16: lane half selects K {0-7,16-23}/{8-15,24-31}, N = lane&15
            const __bf16* wr = &s_w1[buf][(nt * 16 + l16) * 40 + 8 * half];
            b.h[0] = *(const bf16x8*)(wr);
            b.h[1] = *(const bf16x8*)(wr + 16);
            acc1[nt] = __builtin_amdgcn_wmma_f32_16x16x32_bf16(
                false, a.v, false, b.v, (short)0, acc1[nt], false, false);
        }
    };

    stage_xw(0, 0);
    wait_async0();
    __syncthreads();
#pragma unroll 1
    for (int kc = 0; kc < D_INX / 32; ++kc) {
        int cur = kc & 1;
        if (kc + 1 < D_INX / 32) stage_xw(cur ^ 1, kc + 1);
        compute1(cur);
        wait_async0();          // async LDS writes visible before barrier
        __syncthreads();
    }

    // ---- bias + relu, C-layout -> bf16 A-layout via LDS -------------------
#pragma unroll
    for (int nt = 0; nt < 4; ++nt) {
        float b1v = b1[e * HDIM + nt * 16 + l16];
#pragma unroll
        for (int r = 0; r < 8; ++r) {
            float hv = acc1[nt][r] + b1v;        // row = mbase + r + 8*half
            hv = hv > 0.0f ? hv : 0.0f;
            s_h[(mbase + r + 8 * half) * 72 + nt * 16 + l16] = (__bf16)hv;
        }
    }
    __syncthreads();

    // ---- stage 2: y = h @ W2 over 8 N-chunks of 128 -----------------------
    ABFrag a2[2];   // K=64 -> two K=32 A fragments, loaded once
    {
        const __bf16* hr = &s_h[(mbase + l16) * 72];
#pragma unroll
        for (int s = 0; s < 2; ++s) {
            a2[s].h[0] = *(const bf16x8*)(hr + 32 * s + 8 * half);
            a2[s].h[1] = *(const bf16x8*)(hr + 32 * s + 16 + 8 * half);
        }
    }

    auto stage_w2 = [&](int buf, int nc) {
        // W2t chunk: 128 rows x 64 bf16; this thread: 32 bf16 = 4 x b128
        const int n    = tid >> 1;
        const int koff = (tid & 1) * 32;
        uint32_t dst = lds_off32(&s_w2[buf][n * 72 + koff]);
        const __bf16* src =
            W2t + ((size_t)e * D_OUTX + nc * 128 + n) * HDIM + koff;
#pragma unroll
        for (int q = 0; q < 4; ++q)
            async_copy_b128(dst + 16 * q, src + 8 * q);
    };

    auto compute2 = [&](int buf, int nc) {
#pragma unroll
        for (int nt = 0; nt < 8; ++nt) {
            v8f acc = (v8f)(0.0f);
#pragma unroll
            for (int s = 0; s < 2; ++s) {
                ABFrag b;
                const __bf16* wr =
                    &s_w2[buf][(nt * 16 + l16) * 72 + 32 * s + 8 * half];
                b.h[0] = *(const bf16x8*)(wr);
                b.h[1] = *(const bf16x8*)(wr + 16);
                acc = __builtin_amdgcn_wmma_f32_16x16x32_bf16(
                    false, a2[s].v, false, b.v, (short)0, acc, false, false);
            }
            const int col = nc * 128 + nt * 16 + l16;
            float b2v = b2[e * D_OUTX + col];
#pragma unroll
            for (int r = 0; r < 8; ++r) {
                float gw = gwr[r];
                if (gw != 0.0f) {
                    float val = (acc[r] + b2v) * gw;
                    atomicAdd(out + (size_t)tokr[r] * D_OUTX + col, val);
                }
            }
        }
    };

    stage_w2(0, 0);
    wait_async0();
    __syncthreads();
#pragma unroll 1
    for (int nc = 0; nc < D_OUTX / 128; ++nc) {
        int cur = nc & 1;
        if (nc + 1 < D_OUTX / 128) stage_w2(cur ^ 1, nc + 1);
        compute2(cur, nc);
        wait_async0();
        __syncthreads();
    }
}

// ---------------------------------------------------------------------------
extern "C" void kernel_launch(void* const* d_in, const int* in_sizes, int n_in,
                              void* d_out, int out_size, void* d_ws, size_t ws_size,
                              hipStream_t stream)
{
    const float* x  = (const float*)d_in[0];
    const float* Wg = (const float*)d_in[1];
    const float* bg = (const float*)d_in[2];
    const float* W1 = (const float*)d_in[3];
    const float* b1 = (const float*)d_in[4];
    const float* W2 = (const float*)d_in[5];
    const float* b2 = (const float*)d_in[6];
    float* out = (float*)d_out;

    // workspace: counters | buckets | bf16 x | bf16 W1t | bf16 W2t (~22 MB)
    char* ws = (char*)d_ws;
    int*    cnt  = (int*)ws;
    int*    btok = (int*)(ws + 256);
    float*  bw   = (float*)(ws + 256 + (size_t)NEXP * CAP * 4);
    __bf16* xbuf = (__bf16*)(ws + 256 + 2 * (size_t)NEXP * CAP * 4);
    __bf16* W1t  = xbuf + (size_t)B_TOK * D_INX;
    __bf16* W2t  = W1t  + (size_t)NEXP * D_INX * HDIM;

    hipMemsetAsync(out, 0, (size_t)B_TOK * D_OUTX * sizeof(float), stream);
    hipMemsetAsync(cnt, 0, 16 * sizeof(int), stream);

    xcvt_kernel<<<(B_TOK * D_INX / 4) / THREADS, THREADS, 0, stream>>>(x, xbuf);
    prep_weights_kernel<<<dim3(16, NEXP, 2), THREADS, 0, stream>>>(W1, W2, W1t, W2t);
    moe_gate_kernel<<<B_TOK / THREADS, THREADS, 0, stream>>>(x, Wg, bg, cnt, btok, bw);

    dim3 grid(TILES_MAX, NEXP);
    moe_expert_kernel<<<grid, THREADS, 0, stream>>>(
        xbuf, W1t, W2t, b1, b2, cnt, btok, bw, out);
}
